// MultipleAgentsNet_28149215658091
// MI455X (gfx1250) — compile-verified, matrix operand order checked by reading, not run on hardware
//
#include <hip/hip_runtime.h>

// CDNA5 (gfx1250) fused conv3x3(+bias,relu,mask) + per-sample head GEMM.
// One workgroup = (batch, 4 image rows = 128 pixels). 8 waves (wave32),
// each wave owns one 16-pixel M-tile x 128 output channels.
// Conv = 9 shifted GEMMs, K=20 channels padded to 32, v_wmma_f32_16x16x32_bf16.
// Input tile is staged global->LDS with GLOBAL_LOAD_ASYNC_TO_LDS_B128 (ASYNCcnt),
// overlapped with conv-weight B-fragment staging; s_wait_asynccnt gates the
// fp32->bf16 conversion pass.

typedef __attribute__((ext_vector_type(16))) __bf16 v16bf;
typedef __attribute__((ext_vector_type(8)))  float  v8f;

#define BATCH 512
#define IN_CH 20
#define HH 32
#define WW 32
#define FEAT 128
#define NACT 16
#define NUNIT 12

union Frag {
  v16bf v;
  uint4 q[2];
};

__device__ __forceinline__ unsigned short f2bf(float f) {
  unsigned u = __builtin_bit_cast(unsigned, f);
  u += 0x7FFFu + ((u >> 16) & 1u);   // round-to-nearest-even
  return (unsigned short)(u >> 16);
}

__global__ __launch_bounds__(256)
void fused_conv_head(const float* __restrict__ states,
                     const int*   __restrict__ labels,
                     const float* __restrict__ maskings,
                     const float* __restrict__ conv_w,
                     const float* __restrict__ conv_b,
                     const float* __restrict__ W_all,
                     const float* __restrict__ b_all,
                     float* __restrict__ out) {
  // Raw fp32 input tile staged by async DMA: [ci 20][halo row 6][col 32]
  __shared__ __align__(16) float          s_raw[IN_CH * 6 * 32];      // 15360 B
  // Input tile bf16: 6 halo rows x 34 halo cols x 32 channels (20 real + 12 zero pad).
  __shared__ __align__(16) unsigned short s_in[6 * 34 * 32];          // 13056 B
  // Conv weights pre-swizzled into WMMA B-fragment order: [tap][ntile][lane][16]
  __shared__ __align__(16) unsigned short s_b[9 * 8 * 32 * 16];       // 73728 B
  // Post-conv feature tiles (bf16): [128 px][128 feat]
  __shared__ __align__(16) unsigned short s_x[128 * FEAT];            // 32768 B
  // Selected head weights in B-fragment order: [kstep][lane][16]
  __shared__ __align__(16) unsigned short s_w[4 * 32 * 16];           //  4096 B

  const int tid = threadIdx.x;
  const int bat = blockIdx.y;
  const int r0  = blockIdx.x * 4;          // first image row of this block
  const int label = labels[bat];

  // ---- 1) issue async global->LDS copies of the raw input tile (16B chunks) ----
  // 960 chunks of 4 floats: chunk i -> (ci, lrow, col4). Rows outside the image
  // are skipped (s_in is pre-zeroed; conversion also skips them).
  const float* sb = states + (size_t)bat * (IN_CH * HH * WW);
  for (int i = tid; i < IN_CH * 6 * 8; i += 256) {
    int col4 = i & 7;
    int lrow = (i >> 3) % 6;
    int ci   = i / 48;
    int grow = r0 - 1 + lrow;
    if (grow >= 0 && grow < HH) {
      unsigned ldsa = (unsigned)(uintptr_t)&s_raw[(ci * 6 + lrow) * 32 + col4 * 4];
      unsigned long long ga =
          (unsigned long long)(uintptr_t)(sb + ci * (HH * WW) + grow * WW + col4 * 4);
      asm volatile("global_load_async_to_lds_b128 %0, %1, off"
                   :: "v"(ldsa), "v"(ga));
    }
  }

  // ---- 2) zero the bf16 input tile (pad channels / halo borders) ----
  for (int i = tid; i < (6 * 34 * 32) / 2; i += 256) ((unsigned*)s_in)[i] = 0u;

  // ---- 3) stage conv weights into B-fragment layout (overlaps async input DMA) ----
  // fragment elem: lane ln holds col N = ln&15; K = 8*(ln>>4) + q (q<8) or 16+8*(ln>>4)+(q-8)
  for (int i = tid; i < 9 * 8 * 32 * 16; i += 256) {
    int q  = i & 15;
    int ln = (i >> 4) & 31;
    int nt = (i >> 9) & 7;
    int t  = i >> 12;                       // tap = dy*3+dx
    int co  = nt * 16 + (ln & 15);
    int cbs = (ln >> 4) * 8;
    int K   = (q < 8) ? (cbs + q) : (16 + cbs + (q - 8));
    float v = (K < IN_CH) ? conv_w[(co * IN_CH + K) * 9 + t] : 0.f;
    s_b[i] = f2bf(v);
  }

  // ---- 4) stage label-selected head weights as B-fragments (K=feat, N=action) ----
  for (int i = tid; i < 4 * 32 * 16; i += 256) {
    int q  = i & 15;
    int ln = (i >> 4) & 31;
    int s  = i >> 9;
    int k   = ln & 15;
    int cbs = (ln >> 4) * 8;
    int K   = (q < 8) ? (cbs + q) : (16 + cbs + (q - 8));
    int co  = 32 * s + K;
    s_w[i] = f2bf(W_all[((size_t)label * NACT + k) * FEAT + co]);
  }

  __syncthreads();   // separates s_in zeroing from s_in fill; s_b/s_w complete

  // ---- 5) wait for this wave's async copies, then convert own chunks to bf16 ----
  asm volatile("s_wait_asynccnt 0" ::: "memory");
  for (int i = tid; i < IN_CH * 6 * 8; i += 256) {
    int col4 = i & 7;
    int lrow = (i >> 3) % 6;
    int ci   = i / 48;
    int grow = r0 - 1 + lrow;
    if (grow >= 0 && grow < HH) {
      const float4 r = *(const float4*)&s_raw[(ci * 6 + lrow) * 32 + col4 * 4];
      int base = lrow * 34 + col4 * 4 + 1;  // +1 = left halo column
      s_in[(base + 0) * 32 + ci] = f2bf(r.x);
      s_in[(base + 1) * 32 + ci] = f2bf(r.y);
      s_in[(base + 2) * 32 + ci] = f2bf(r.z);
      s_in[(base + 3) * 32 + ci] = f2bf(r.w);
    }
  }
  __syncthreads();   // s_in fully built before cross-thread WMMA reads

  const int lane = tid & 31;
  const int wv   = tid >> 5;       // wave = M-tile index (0..7)
  const int lm   = lane & 15;
  const int hs   = lane >> 4;      // lane half
  const int cb   = hs * 8;         // K base for A/B fragments

  // ---- conv GEMM: acc[nt] = bias; acc += sum over 9 taps A(16x32) * B(32x16) ----
  v8f acc[8];
  #pragma unroll
  for (int nt = 0; nt < 8; ++nt) {
    float bv = conv_b[nt * 16 + lm];                 // C-layout: N = lane&15 (same all rows)
    v8f t8 = {bv, bv, bv, bv, bv, bv, bv, bv};
    acc[nt] = t8;
  }

  const int p  = wv * 16 + lm;     // A-row pixel within 128-pixel block (row = lane&15)
  const int ry = p >> 5;
  const int cx = p & 31;

  #pragma unroll
  for (int t = 0; t < 9; ++t) {
    const int dyp = t / 3, dxp = t % 3;
    const int pos = (ry + dyp) * 34 + (cx + dxp);    // halo coords, always in-bounds
    Frag a;
    a.q[0] = *(const uint4*)&s_in[pos * 32 + cb];
    a.q[1] = *(const uint4*)&s_in[pos * 32 + cb + 16];
    #pragma unroll
    for (int nt = 0; nt < 8; ++nt) {
      Frag bm;
      const unsigned short* bp = &s_b[((t * 8 + nt) * 32 + lane) * 16];
      bm.q[0] = *(const uint4*)bp;
      bm.q[1] = *(const uint4*)(bp + 8);
      acc[nt] = __builtin_amdgcn_wmma_f32_16x16x32_bf16(
          false, a.v, false, bm.v, (short)0, acc[nt], false, false);
    }
  }

  // ---- epilogue: relu * mask, write bf16 X tile (transpose via LDS for head A-frags) ----
  const float* mb  = maskings + (size_t)bat * (HH * WW);
  const int pixb = blockIdx.x * 128;
  #pragma unroll
  for (int nt = 0; nt < 8; ++nt) {
    int co = nt * 16 + lm;
    #pragma unroll
    for (int v = 0; v < 8; ++v) {
      int px = v + 8 * hs;                 // C-layout row
      int tp = wv * 16 + px;
      float x = acc[nt][v];
      x = fmaxf(x, 0.f) * mb[pixb + tp];
      s_x[tp * FEAT + co] = f2bf(x);
    }
  }
  // Only this wave reads its own X region; same-wave DS ordering suffices.

  // ---- head GEMM: Out(16x16) = X(16x128) * Wsel^T(128x16) + b_sel ----
  v8f h;
  {
    float bs = b_all[label * NACT + lm];
    v8f t8 = {bs, bs, bs, bs, bs, bs, bs, bs};
    h = t8;
  }
  #pragma unroll
  for (int s = 0; s < 4; ++s) {
    Frag a, bm;
    const unsigned short* xp = &s_x[(wv * 16 + lm) * FEAT + 32 * s + cb];
    a.q[0] = *(const uint4*)xp;
    a.q[1] = *(const uint4*)(xp + 16);
    const unsigned short* wp = &s_w[(s * 32 + lane) * 16];
    bm.q[0] = *(const uint4*)wp;
    bm.q[1] = *(const uint4*)(wp + 8);
    h = __builtin_amdgcn_wmma_f32_16x16x32_bf16(
        false, a.v, false, bm.v, (short)0, h, false, false);
  }

  // ---- write split outputs: part1 [B,32,32,12], part2 [B,32,32,4] concatenated ----
  float* out2 = out + (size_t)BATCH * HH * WW * NUNIT;
  const int k = lm;
  #pragma unroll
  for (int v = 0; v < 8; ++v) {
    int gp = pixb + wv * 16 + v + 8 * hs;
    size_t base = (size_t)bat * (HH * WW) + gp;
    float val = h[v];
    if (k < NUNIT) out[base * NUNIT + k] = val;
    else           out2[base * 4 + (k - NUNIT)] = val;
  }
}

extern "C" void kernel_launch(void* const* d_in, const int* in_sizes, int n_in,
                              void* d_out, int out_size, void* d_ws, size_t ws_size,
                              hipStream_t stream) {
  const float* states   = (const float*)d_in[0];
  const int*   labels   = (const int*)  d_in[1];
  const float* maskings = (const float*)d_in[2];
  const float* conv_w   = (const float*)d_in[3];
  const float* conv_b   = (const float*)d_in[4];
  const float* W_all    = (const float*)d_in[5];
  const float* b_all    = (const float*)d_in[6];
  float* out = (float*)d_out;
  (void)in_sizes; (void)n_in; (void)out_size; (void)d_ws; (void)ws_size;

  dim3 grid(8, BATCH);   // 8 row-blocks of 4 rows x 512 batches = 4096 WGs
  dim3 block(256);       // 8 wave32 waves
  hipLaunchKernelGGL(fused_conv_head, grid, block, 0, stream,
                     states, labels, maskings, conv_w, conv_b, W_all, b_all, out);
}
